// MultiHeadAttention_80659485819175
// MI455X (gfx1250) — matrix operand order repaired and checked by reading, not compiled, not run on hardware
//
#include <hip/hip_runtime.h>
#include <hip/hip_bf16.h>

// MultiHeadAttention (relu-normalized, masked) for MI455X / gfx1250.
// B=16, U=1024, HEADS=8, S=1024, C=128.
// Pipeline: rowcount -> QKV gemm (bf16 wmma) -> fused attention (bf16 wmma x2)
//           -> output gemm (bf16 wmma, fp32 out).

#define B_DIM 16
#define U_DIM 1024
#define H_DIM 8
#define S_DIM 1024
#define C_DIM 128  // U/H

// Optional CDNA5 async global->LDS path (guarded so either toolchain compiles).
#if defined(__has_builtin)
#  if __has_builtin(__builtin_amdgcn_global_load_async_to_lds_b128) && \
      __has_builtin(__builtin_amdgcn_s_wait_asynccnt)
#    define USE_ASYNC_LDS 1
#  endif
#endif
#ifndef USE_ASYNC_LDS
#  define USE_ASYNC_LDS 0
#endif

typedef __attribute__((ext_vector_type(16))) __bf16 v16bf;
typedef __attribute__((ext_vector_type(8)))  float  v8f;
typedef int v4i_vec __attribute__((vector_size(4 * sizeof(int))));
typedef __attribute__((address_space(1))) v4i_vec* gptr_v4i;
typedef __attribute__((address_space(3))) v4i_vec* lptr_v4i;

union V16B { v16bf v; unsigned int u[8]; };

__device__ __forceinline__ unsigned short f2bf(float f) {
  union { float f; unsigned int u; } x; x.f = f;
  unsigned int u = x.u;
  unsigned int r = u + 0x7FFFu + ((u >> 16) & 1u);  // round-to-nearest-even
  return (unsigned short)(r >> 16);
}

// A-operand (16x32 bf16, MxK): lane half selects K-octet; pairs consecutive in K.
__device__ __forceinline__ int a_kbase(int i, int half) {
  return (i < 4 ? 2 * i : 16 + 2 * (i - 4)) + half * 8;
}
// B-operand (32x16 bf16, KxN): lane half selects K-octet (same as A); pairs consecutive in K.
__device__ __forceinline__ int b_kbase(int i, int half) {
  return (i < 4 ? 2 * i : 16 + 2 * (i - 4)) + half * 8;
}

// ---------------------------------------------------------------------------
// Kernel 1: per-row mask counts -> reciprocal (float)
// mask: (B,S,S) bool bytes. rcnt[b*S+q] = 1/max(sum_k mask[b,q,k],1)
// ---------------------------------------------------------------------------
__global__ void rowcount_kernel(const unsigned char* __restrict__ mask,
                                float* __restrict__ rcnt) {
  int row  = blockIdx.x * 8 + (threadIdx.x >> 5);   // 8 waves per block
  int lane = threadIdx.x & 31;
  const uint4* p4 = (const uint4*)(mask + (size_t)row * S_DIM);
  int sum = 0;
  for (int i = lane; i < S_DIM / 16; i += 32) {     // 64 uint4 per row
    uint4 v = p4[i];
    sum += __popc(v.x) + __popc(v.y) + __popc(v.z) + __popc(v.w);
  }
  for (int off = 16; off > 0; off >>= 1) sum += __shfl_down(sum, off, 32);
  if (lane == 0) rcnt[row] = 1.0f / (float)(sum > 0 ? sum : 1);
}

// ---------------------------------------------------------------------------
// Kernel 2/4: batched GEMM  Out[z] (MxN) = W (MxK, fp32) * X[z] (KxN)
// X is fp32 (XBF=0) or bf16-as-ushort (XBF=1); Out is bf16 (OBF=1) or fp32.
// Block: 256 threads (8 waves). Tile: 128(M) x 64(N), K-step 64 (2x WMMA-K).
// ---------------------------------------------------------------------------
template <int XBF, int OBF>
__global__ void gemm_wmma_kernel(const float* __restrict__ W,
                                 const void* __restrict__ Xv,
                                 void* __restrict__ Outv,
                                 int M, int K, int N) {
  __shared__ unsigned short Wt[128 * 66];  // [m][k], 64-wide K slab, pad 2
  __shared__ unsigned short Xt[64 * 66];   // [n][k], transposed for B-op

  const int tid  = threadIdx.x;
  const int wave = tid >> 5;
  const int lane = tid & 31;
  const int half = lane >> 4;
  const int lr   = lane & 15;

  const int n0 = blockIdx.x * 64;
  const int m0 = blockIdx.y * 128;
  const size_t zoffX = (size_t)blockIdx.z * K * N;
  const size_t zoffO = (size_t)blockIdx.z * M * N;

  const float*          Xf = (const float*)Xv + zoffX;
  const unsigned short* Xb = (const unsigned short*)Xv + zoffX;
  float*                Of = (float*)Outv + zoffO;
  unsigned short*       Ob = (unsigned short*)Outv + zoffO;

  v8f acc0 = {}, acc1 = {}, acc2 = {}, acc3 = {};

  const int ksteps = K >> 6;  // 64-wide K slabs
  for (int kt = 0; kt < ksteps; ++kt) {
    const int kbase = kt << 6;
    // Load W slab 128x64 (fp32 -> bf16), coalesced along k.
    for (int e = tid; e < 128 * 64; e += 256) {
      int m = e >> 6, kk = e & 63;
      Wt[m * 66 + kk] = f2bf(W[(size_t)(m0 + m) * K + kbase + kk]);
    }
    // Load X slab 64(k) x 64(n), store transposed [n][k]; coalesced along n.
    for (int e = tid; e < 64 * 64; e += 256) {
      int n = e & 63, kk = e >> 6;
      if (XBF)
        Xt[n * 66 + kk] = Xb[(size_t)(kbase + kk) * N + n0 + n];
      else
        Xt[n * 66 + kk] = f2bf(Xf[(size_t)(kbase + kk) * N + n0 + n]);
    }
    // Prefetch next K-slab into caches.
    if (kt + 1 < ksteps) {
      __builtin_prefetch(&W[(size_t)(m0 + (tid >> 1)) * K + kbase + 64], 0, 1);
      if (XBF)
        __builtin_prefetch(&Xb[(size_t)(kbase + 64 + (tid >> 2)) * N + n0], 0, 1);
      else
        __builtin_prefetch(&Xf[(size_t)(kbase + 64 + (tid >> 2)) * N + n0], 0, 1);
    }
    __syncthreads();

#pragma unroll
    for (int c0 = 0; c0 < 64; c0 += 32) {
      // A operand: rows m = wave*16 + lr
      V16B a;
#pragma unroll
      for (int i = 0; i < 8; ++i)
        a.u[i] = *(const unsigned int*)
            &Wt[(wave * 16 + lr) * 66 + c0 + a_kbase(i, half)];

#pragma unroll
      for (int ns = 0; ns < 4; ++ns) {
        V16B b;
#pragma unroll
        for (int i = 0; i < 8; ++i)
          b.u[i] = *(const unsigned int*)
              &Xt[(ns * 16 + lr) * 66 + c0 + b_kbase(i, half)];
        v8f* accp = ns == 0 ? &acc0 : ns == 1 ? &acc1 : ns == 2 ? &acc2 : &acc3;
        *accp = __builtin_amdgcn_wmma_f32_16x16x32_bf16(
            false, a.v, false, b.v, (short)0, *accp, false, false);
      }
    }
    __syncthreads();
  }

  // Epilogue: D layout -> row = wave*16 + j + 8*half, col = n0 + ns*16 + lr
#pragma unroll
  for (int ns = 0; ns < 4; ++ns) {
    v8f* accp = ns == 0 ? &acc0 : ns == 1 ? &acc1 : ns == 2 ? &acc2 : &acc3;
#pragma unroll
    for (int j = 0; j < 8; ++j) {
      int m = m0 + wave * 16 + j + 8 * half;
      int n = n0 + ns * 16 + lr;
      float v = (*accp)[j];
      if (OBF) Ob[(size_t)m * N + n] = f2bf(v);
      else     Of[(size_t)m * N + n] = v;
    }
  }
}

// ---------------------------------------------------------------------------
// Kernel 3: fused masked-relu attention per (b, h, 64-wide q tile).
// qkv: bf16 (B, 3U, S); computes Cc[b, h*C+c, q] = sum_k P[q,k] * V[c,k]
// with P = mask ? relu(QK)*scale*rcnt[q] : 0, streamed over 64-wide k tiles.
// ---------------------------------------------------------------------------
__global__ void attn_kernel(const unsigned short* __restrict__ qkv,
                            const unsigned char* __restrict__ mask,
                            const float* __restrict__ rcnt,
                            unsigned short* __restrict__ cc) {
  __shared__ unsigned short Qt[64 * 132];   // [q][c] (A-op tile 1), pad 4
  __shared__ unsigned short Kt[64 * 132];   // [k][c] (B-op tile 1)
  __shared__ unsigned short Vt[128 * 72];   // [c][k] (A-op tile 2), 16B-aligned rows
  __shared__ unsigned short Pt[64 * 68];    // [q][k] (B-op tile 2)
  __shared__ float          Rq[64];         // per-q reciprocal row counts

  const int tid  = threadIdx.x;
  const int wave = tid >> 5;
  const int lane = tid & 31;
  const int half = lane >> 4;
  const int lr   = lane & 15;

  const int q0 = blockIdx.x * 64;
  const int h  = blockIdx.y;
  const int b  = blockIdx.z;
  const float scale = 0.08838834764831845f;  // 1/sqrt(128)

  const unsigned short* Qg = qkv + ((size_t)b * 3 * U_DIM + h * C_DIM) * S_DIM;
  const unsigned short* Kg = Qg + (size_t)U_DIM * S_DIM;
  const unsigned short* Vg = Qg + (size_t)2 * U_DIM * S_DIM;
  const unsigned char*  Mg = mask + (size_t)b * S_DIM * S_DIM;

  // Load Q^T tile [64 q][128 c]; coalesced along q.
  for (int e = tid; e < 64 * 128; e += 256) {
    int q = e & 63, c = e >> 6;
    Qt[q * 132 + c] = Qg[(size_t)c * S_DIM + q0 + q];
  }
  if (tid < 64) Rq[tid] = rcnt[(size_t)b * S_DIM + q0 + tid];
  __syncthreads();

  v8f accv0 = {}, accv1 = {}, accv2 = {}, accv3 = {};

  const int qsub  = wave & 3;
  const int kpair = wave >> 2;

  for (int kt = 0; kt < S_DIM / 64; ++kt) {
    const int k0 = kt * 64;
    // Load K tile [64 k][128 c] (transpose in LDS); coalesced along s.
    for (int e = tid; e < 64 * 128; e += 256) {
      int k = e & 63, c = e >> 6;
      Kt[k * 132 + c] = Kg[(size_t)c * S_DIM + k0 + k];
    }
#if USE_ASYNC_LDS
    // V tile is layout-preserving: async DMA global -> LDS, 16B per lane.
    // 128 rows(c) x 64 elems(k) bf16 = 1024 x 16B chunks; 4 rounds of 256.
    {
#pragma unroll
      for (int r = 0; r < 4; ++r) {
        int e  = tid + r * 256;      // chunk index
        int ch = e & 7, c = e >> 3;  // 8 chunks of 8 elems per c-row
        unsigned short* gsrc =
            const_cast<unsigned short*>(Vg + (size_t)c * S_DIM + k0 + ch * 8);
        __builtin_amdgcn_global_load_async_to_lds_b128(
            (gptr_v4i)gsrc, (lptr_v4i)(&Vt[c * 72 + ch * 8]), 0, 0);
      }
      __builtin_amdgcn_s_wait_asynccnt(0);
    }
#else
    for (int e = tid; e < 128 * 64; e += 256) {
      int k = e & 63, c = e >> 6;
      Vt[c * 72 + k] = Vg[(size_t)c * S_DIM + k0 + k];
    }
#endif
    __syncthreads();

    // ---- WMMA #1: P[64q][64k] = Q^T K ; each wave does 2 of 16 tiles ----
#pragma unroll
    for (int t = 0; t < 2; ++t) {
      const int ksub = kpair * 2 + t;
      v8f acc = {};
#pragma unroll
      for (int c0 = 0; c0 < 128; c0 += 32) {
        V16B a, bb;
#pragma unroll
        for (int i = 0; i < 8; ++i)
          a.u[i] = *(const unsigned int*)
              &Qt[(qsub * 16 + lr) * 132 + c0 + a_kbase(i, half)];
#pragma unroll
        for (int i = 0; i < 8; ++i)
          bb.u[i] = *(const unsigned int*)
              &Kt[(ksub * 16 + lr) * 132 + c0 + b_kbase(i, half)];
        acc = __builtin_amdgcn_wmma_f32_16x16x32_bf16(
            false, a.v, false, bb.v, (short)0, acc, false, false);
      }
      // mask + relu + scale + rownorm, write P tile (bf16) to LDS
#pragma unroll
      for (int j = 0; j < 8; ++j) {
        int ql = qsub * 16 + j + 8 * half;
        int kl = ksub * 16 + lr;
        int qg = q0 + ql, kg = k0 + kl;
        float p = acc[j];
        float r = (Mg[(size_t)qg * S_DIM + kg] != 0)
                      ? fmaxf(p, 0.0f) * scale * Rq[ql] : 0.0f;
        Pt[ql * 68 + kl] = f2bf(r);
      }
    }
    __syncthreads();

    // ---- WMMA #2: Ch[c][q] += V * P^T ; wave owns 16 c-rows x 64 q ----
#pragma unroll
    for (int ns = 0; ns < 4; ++ns) {
      v8f* accp = ns == 0 ? &accv0 : ns == 1 ? &accv1 : ns == 2 ? &accv2 : &accv3;
#pragma unroll
      for (int kk0 = 0; kk0 < 64; kk0 += 32) {
        V16B a, bb;
#pragma unroll
        for (int i = 0; i < 8; ++i)
          a.u[i] = *(const unsigned int*)
              &Vt[(wave * 16 + lr) * 72 + kk0 + a_kbase(i, half)];
#pragma unroll
        for (int i = 0; i < 8; ++i)
          bb.u[i] = *(const unsigned int*)
              &Pt[(ns * 16 + lr) * 68 + kk0 + b_kbase(i, half)];
        *accp = __builtin_amdgcn_wmma_f32_16x16x32_bf16(
            false, a.v, false, bb.v, (short)0, *accp, false, false);
      }
    }
    __syncthreads();
  }

  // Epilogue: Cc[b, h*C + c, q0+q] (bf16)
#pragma unroll
  for (int ns = 0; ns < 4; ++ns) {
    v8f* accp = ns == 0 ? &accv0 : ns == 1 ? &accv1 : ns == 2 ? &accv2 : &accv3;
#pragma unroll
    for (int j = 0; j < 8; ++j) {
      int c = wave * 16 + j + 8 * half;
      int q = q0 + ns * 16 + lr;
      cc[((size_t)b * U_DIM + h * C_DIM + c) * S_DIM + q] = f2bf((*accp)[j]);
    }
  }
}

// ---------------------------------------------------------------------------
extern "C" void kernel_launch(void* const* d_in, const int* in_sizes, int n_in,
                              void* d_out, int out_size, void* d_ws, size_t ws_size,
                              hipStream_t stream) {
  const float*         x     = (const float*)d_in[0];          // (B,U,S) f32
  const unsigned char* mask  = (const unsigned char*)d_in[1];  // (B,S,S) bool
  const float*         w_qkv = (const float*)d_in[2];          // (3U,U) f32
  const float*         w_out = (const float*)d_in[3];          // (U,U) f32
  float*               out   = (float*)d_out;                  // (B,U,S) f32

  // Workspace layout
  unsigned short* qkv_bf = (unsigned short*)d_ws;                       // 3*B*U*S bf16
  unsigned short* cc_bf  = qkv_bf + (size_t)3 * B_DIM * U_DIM * S_DIM;  // B*U*S bf16
  float*          rcnt   = (float*)(cc_bf + (size_t)B_DIM * U_DIM * S_DIM); // B*S f32

  // 1) mask row counts
  rowcount_kernel<<<B_DIM * S_DIM / 8, 256, 0, stream>>>(mask, rcnt);

  // 2) QKV projection: (3U x U) * (U x S) per batch -> bf16
  gemm_wmma_kernel<0, 1><<<dim3(S_DIM / 64, 3 * U_DIM / 128, B_DIM), 256, 0, stream>>>(
      w_qkv, (const void*)x, (void*)qkv_bf, 3 * U_DIM, U_DIM, S_DIM);

  // 3) fused masked attention -> bf16 context
  attn_kernel<<<dim3(S_DIM / 64, H_DIM, B_DIM), 256, 0, stream>>>(
      qkv_bf, mask, rcnt, cc_bf);

  // 4) output projection: (U x U) * (U x S) per batch -> f32
  gemm_wmma_kernel<1, 0><<<dim3(S_DIM / 64, U_DIM / 128, B_DIM), 256, 0, stream>>>(
      w_out, (const void*)cc_bf, (void*)out, U_DIM, U_DIM, S_DIM);
}
